// AttentionScorer_34857954574439
// MI455X (gfx1250) — compile-verified, matrix operand order checked by reading, not run on hardware
//
#include <hip/hip_runtime.h>
#include <hip/hip_bf16.h>

#define Bsz 16
#define Lt  1024
#define Lv  4096
#define Dd  512

// padded LDS row pitch for bf16 rows of 512: 1040 bytes = 260 dwords == 4 (mod 64)
// -> 16 lanes reading b128 at stride 1040B hit disjoint bank groups.
#define PITCH 1040

typedef __attribute__((ext_vector_type(16))) __bf16 v16bf;
typedef __attribute__((ext_vector_type(8)))  float  v8f;

union ABFrag { uint4 u[2]; v16bf v; };

// async global -> LDS copy of one 16-byte chunk (per-lane), ASYNCcnt-tracked
__device__ __forceinline__ void async_cp16(unsigned lds_off, const void* gsrc) {
    unsigned long long ga = (unsigned long long)(size_t)gsrc;
    asm volatile("global_load_async_to_lds_b128 %0, %1, off"
                 :: "v"(lds_off), "v"(ga) : "memory");
}
__device__ __forceinline__ void wait_async_le(int n) {
    if (n == 0) asm volatile("s_wait_asynccnt 0x0" ::: "memory");
    else        asm volatile("s_wait_asynccnt 0x4" ::: "memory");
}

// float -> bf16 round-to-nearest-even
__device__ __forceinline__ unsigned short f2bf(float f) {
    unsigned int u = __float_as_uint(f);
    if ((u & 0x7fffffffu) > 0x7f800000u) return (unsigned short)0x7fc0; // NaN
    u += 0x7fffu + ((u >> 16) & 1u);
    return (unsigned short)(u >> 16);
}

// ---------------------------------------------------------------------------
// K1: nan_to_num + L2 normalize along D, emit bf16 rows. One wave per row.
// ---------------------------------------------------------------------------
__global__ void normalize_kernel(const float* __restrict__ in,
                                 unsigned short* __restrict__ out, int nrows) {
    int row  = blockIdx.x * 8 + (threadIdx.x >> 5);
    int lane = threadIdx.x & 31;
    if (row >= nrows) return;
    const float* p = in + (size_t)row * Dd + lane * 16;
    float x[16];
    float ss = 0.f;
#pragma unroll
    for (int i = 0; i < 4; ++i) {
        float4 f = ((const float4*)p)[i];
        float t4[4] = {f.x, f.y, f.z, f.w};
#pragma unroll
        for (int j = 0; j < 4; ++j) {
            float xv = t4[j];
            if (xv != xv) xv = 0.f;                             // nan -> 0
            else if (fabsf(xv) > 3.4028234e38f)                 // inf -> +/- max
                xv = copysignf(3.4028234e38f, xv);
            x[i * 4 + j] = xv;
            ss += xv * xv;
        }
    }
#pragma unroll
    for (int off = 16; off; off >>= 1) ss += __shfl_xor(ss, off, 32);
    float scale = 1.0f / fmaxf(sqrtf(ss), 1e-6f);
    unsigned short o[16];
#pragma unroll
    for (int i = 0; i < 16; ++i) o[i] = f2bf(x[i] * scale);
    uint4* dst = (uint4*)(out + (size_t)row * Dd + lane * 16);
    dst[0] = ((const uint4*)o)[0];
    dst[1] = ((const uint4*)o)[1];
}

// ---------------------------------------------------------------------------
// K2 (GEMM sweep 1): workgroup = batch b, 32 vision rows x all 1024 text cols.
// A (32 rows) staged in LDS via async copies; B tiles read from global (L2).
// exp(S/TEMP) tile buffered in LDS; emits text_acc and colsum atomically.
// LDS: Sbuf 128KB + rowinv + Astage 32.5KB ~= 161 KB
// ---------------------------------------------------------------------------
__global__ void __launch_bounds__(256)
pass1_kernel(const unsigned short* __restrict__ tn,
             const unsigned short* __restrict__ vn,
             float* __restrict__ text_acc,
             float* __restrict__ colsum) {
    extern __shared__ char smem[];
    float* Sbuf   = (float*)smem;                       // [32][1024] exp(S*0.5)
    float* rowinv = (float*)(smem + 32 * 1024 * 4);     // [32]
    char*  Abuf   = smem + 32 * 1024 * 4 + 128;         // 32 rows, PITCH bytes each

    const int b     = blockIdx.y;
    const int vbase = blockIdx.x * 32;
    const int tid   = threadIdx.x;
    const int wave  = tid >> 5;
    const int lane  = tid & 31;
    const int lid   = lane & 15;

    const unsigned short* vb = vn + (size_t)b * Lv * Dd;
    const unsigned short* tb = tn + (size_t)b * Lt * Dd;

    // stage A: 32 rows x 64 chunks of 16B (async), 8 chunks per thread
    for (int i = tid; i < 32 * 64; i += 256) {
        int row = i >> 6, ch = i & 63;
        async_cp16((unsigned)(size_t)(Abuf + row * PITCH + ch * 16),
                   vb + (size_t)(vbase + row) * Dd + ch * 8);
    }
    wait_async_le(0);
    __syncthreads();

    const int kselB = (lane >> 4) * 16;   // byte offset selecting K {0..7} vs {8..15}
    // 2 M-tiles x 64 N-tiles = 128 tiles, 16 per wave (wave<4 -> mt0, wave>=4 -> mt1)
    for (int i = 0; i < 16; ++i) {
        asm volatile("" ::: "memory");    // block cross-iteration operand hoisting
        int tile = wave * 16 + i;
        int mt = tile >> 6;
        int nt = tile & 63;
        const char* Arow = Abuf + (mt * 16 + lid) * PITCH + kselB;
        const unsigned short* bp = tb + (size_t)(nt * 16 + lid) * Dd + (kselB >> 1);
        v8f c = {0.f, 0.f, 0.f, 0.f, 0.f, 0.f, 0.f, 0.f};
#pragma unroll
        for (int k0 = 0; k0 < Dd; k0 += 32) {
            ABFrag a, bfr;
            a.u[0]   = *(const uint4*)(Arow + k0 * 2);        // ds_load_b128
            a.u[1]   = *(const uint4*)(Arow + k0 * 2 + 32);
            bfr.u[0] = *(const uint4*)(bp + k0);              // global b128
            bfr.u[1] = *(const uint4*)(bp + k0 + 16);
            c = __builtin_amdgcn_wmma_f32_16x16x32_bf16(
                    false, a.v, false, bfr.v, (short)0, c, false, false);
        }
        // C layout: VGPR j -> local row (lane<16 ? j : j+8), col = lane&15
        int rbase = mt * 16 + ((lane >> 4) << 3);
        int col   = nt * 16 + lid;
#pragma unroll
        for (int j = 0; j < 8; ++j)
            Sbuf[(rbase + j) * 1024 + col] = __expf(0.5f * c[j]);
    }
    __syncthreads();

    // per-row softmax denominators (rows wave*4 .. wave*4+3)
    for (int r = wave * 4; r < wave * 4 + 4; ++r) {
        float s = 0.f;
        for (int cidx = lane; cidx < 1024; cidx += 32) s += Sbuf[r * 1024 + cidx];
#pragma unroll
        for (int off = 16; off; off >>= 1) s += __shfl_xor(s, off, 32);
        if (lane == 0) rowinv[r] = 1.0f / s;
    }
    __syncthreads();

    // per-column accumulation over the 32 rows of this tile
    for (int col = tid; col < 1024; col += 256) {
        float ta = 0.f, cs = 0.f;
#pragma unroll 4
        for (int r = 0; r < 32; ++r) {
            float e = Sbuf[r * 1024 + col];
            cs += e;
            ta += e * rowinv[r];
        }
        atomicAdd(&text_acc[b * Lt + col], ta);
        atomicAdd(&colsum[b * Lt + col], cs);
    }
}

// ---------------------------------------------------------------------------
// K3: per batch: threshold mask over text tokens, fold mask/cnt/colsum into
// coeff[b,t] = mask / (cnt * colsum).
// ---------------------------------------------------------------------------
__global__ void select_kernel(const float* __restrict__ text_acc,
                              const float* __restrict__ colsum,
                              float* __restrict__ coeff) {
    __shared__ float sred[256];
    const int b = blockIdx.x, tid = threadIdx.x;
    const float* ta = text_acc + b * Lt;

    float local = 0.f;
    for (int t = tid; t < Lt; t += 256) local += ta[t];
    sred[tid] = local; __syncthreads();
    for (int s = 128; s; s >>= 1) { if (tid < s) sred[tid] += sred[tid + s]; __syncthreads(); }
    const float TH = sred[0] / (float)Lt;
    __syncthreads();

    float lc = 0.f;
    for (int t = tid; t < Lt; t += 256) if (ta[t] >= TH) lc += 1.f;
    sred[tid] = lc; __syncthreads();
    for (int s = 128; s; s >>= 1) { if (tid < s) sred[tid] += sred[tid + s]; __syncthreads(); }
    const float cntf = sred[0];
    const bool  fallback = (cntf < 0.5f);
    const float cnt = fallback ? (float)Lt : cntf;

    for (int t = tid; t < Lt; t += 256) {
        bool m = fallback || (ta[t] >= TH);
        coeff[b * Lt + t] = m ? 1.0f / (cnt * colsum[b * Lt + t]) : 0.0f;
    }
}

// ---------------------------------------------------------------------------
// K4 (GEMM sweep 2): workgroup = 128 vision rows (8 waves x 16) x all Lt.
// A (128 rows) staged once in LDS (async). B tiles (16 t-rows) double-buffered
// in LDS via async copies shared by all 8 waves. vision[v] computed wave-locally.
// LDS: Astage 130KB + 2 B buffers 32.5KB ~= 163 KB
// ---------------------------------------------------------------------------
__global__ void __launch_bounds__(256)
pass2_kernel(const unsigned short* __restrict__ tn,
             const unsigned short* __restrict__ vn,
             const float* __restrict__ coeff,
             float* __restrict__ vision) {
    extern __shared__ char smem[];
    char* Abuf = smem;                                // 128 rows * PITCH
    char* Bbuf0 = smem + 128 * PITCH;                 // 16 rows * PITCH
    char* Bbuf1 = Bbuf0 + 16 * PITCH;

    const int b     = blockIdx.y;
    const int tid   = threadIdx.x;
    const int wave  = tid >> 5;
    const int lane  = tid & 31;
    const int lid   = lane & 15;
    const int vbase = blockIdx.x * 128;

    const unsigned short* va = vn + ((size_t)b * Lv + vbase) * Dd;
    const unsigned short* tb = tn + (size_t)b * Lt * Dd;
    const float* cf = coeff + b * Lt;

    // stage A: 128 rows x 64 chunks = 8192 chunks, 32 per thread (async)
    for (int i = tid; i < 128 * 64; i += 256) {
        int row = i >> 6, ch = i & 63;
        async_cp16((unsigned)(size_t)(Abuf + row * PITCH + ch * 16),
                   va + (size_t)row * Dd + ch * 8);
    }
    // preload B tile 0 (1024 chunks, 4 per thread)
    {
        const unsigned short* src = tb;
        for (int i = tid; i < 1024; i += 256) {
            int row = i >> 6, ch = i & 63;
            async_cp16((unsigned)(size_t)(Bbuf0 + row * PITCH + ch * 16),
                       src + (size_t)row * Dd + ch * 8);
        }
    }
    wait_async_le(0);
    __syncthreads();

    const int kselB = (lane >> 4) * 16;
    const char* Arow = Abuf + (wave * 16 + lid) * PITCH + kselB;
    float vsum[8] = {0.f, 0.f, 0.f, 0.f, 0.f, 0.f, 0.f, 0.f};

    for (int nt = 0; nt < 64; ++nt) {
        char* curB = (nt & 1) ? Bbuf1 : Bbuf0;
        char* nxtB = (nt & 1) ? Bbuf0 : Bbuf1;
        if (nt + 1 < 64) {
            const unsigned short* src = tb + (size_t)(nt + 1) * 16 * Dd;
            for (int i = tid; i < 1024; i += 256) {
                int row = i >> 6, ch = i & 63;
                async_cp16((unsigned)(size_t)(nxtB + row * PITCH + ch * 16),
                           src + (size_t)row * Dd + ch * 8);
            }
            wait_async_le(4);   // oldest (current tile's) copies complete
        } else {
            wait_async_le(0);
        }
        __syncthreads();        // B(nt) visible to all waves; also blocks hoisting

        const char* Brow = curB + lid * PITCH + kselB;
        v8f c = {0.f, 0.f, 0.f, 0.f, 0.f, 0.f, 0.f, 0.f};
#pragma unroll
        for (int k0 = 0; k0 < Dd; k0 += 32) {
            ABFrag a, bfr;
            a.u[0]   = *(const uint4*)(Arow + k0 * 2);
            a.u[1]   = *(const uint4*)(Arow + k0 * 2 + 32);
            bfr.u[0] = *(const uint4*)(Brow + k0 * 2);
            bfr.u[1] = *(const uint4*)(Brow + k0 * 2 + 32);
            c = __builtin_amdgcn_wmma_f32_16x16x32_bf16(
                    false, a.v, false, bfr.v, (short)0, c, false, false);
        }
        float w = cf[nt * 16 + lid];
#pragma unroll
        for (int j = 0; j < 8; ++j) vsum[j] += w * __expf(0.5f * c[j]);
        __syncthreads();        // done reading curB before it is overwritten
    }

    // reduce over N: 16-lane groups (lanes 0-15 -> rows 0..7, 16-31 -> rows 8..15)
#pragma unroll
    for (int off = 1; off < 16; off <<= 1)
#pragma unroll
        for (int j = 0; j < 8; ++j) vsum[j] += __shfl_xor(vsum[j], off, 32);

    float* outp = vision + (size_t)b * Lv + vbase + wave * 16;
    if (lane == 0)  { for (int j = 0; j < 8; ++j) outp[j]     = vsum[j]; }
    if (lane == 16) { for (int j = 0; j < 8; ++j) outp[8 + j] = vsum[j]; }
}

// ---------------------------------------------------------------------------
// K5: per batch min-max normalization of vision scores.
// ---------------------------------------------------------------------------
__global__ void minmax_kernel(const float* __restrict__ vision,
                              float* __restrict__ out) {
    __shared__ float smn[256], smx[256];
    const int b = blockIdx.x, tid = threadIdx.x;
    const float* vb = vision + (size_t)b * Lv;
    float mn = 3.4028235e38f, mx = -3.4028235e38f;
    for (int v = tid; v < Lv; v += 256) {
        float x = vb[v];
        mn = fminf(mn, x); mx = fmaxf(mx, x);
    }
    smn[tid] = mn; smx[tid] = mx; __syncthreads();
    for (int s = 128; s; s >>= 1) {
        if (tid < s) {
            smn[tid] = fminf(smn[tid], smn[tid + s]);
            smx[tid] = fmaxf(smx[tid], smx[tid + s]);
        }
        __syncthreads();
    }
    const float lo = smn[0];
    const float scale = 1.0f / (smx[0] - smn[0] + 1e-6f);
    for (int v = tid; v < Lv; v += 256)
        out[(size_t)b * Lv + v] = (vb[v] - lo) * scale;
}

// ---------------------------------------------------------------------------
extern "C" void kernel_launch(void* const* d_in, const int* in_sizes, int n_in,
                              void* d_out, int out_size, void* d_ws, size_t ws_size,
                              hipStream_t stream) {
    const float* t = (const float*)d_in[0];   // [B, Lt, D] f32
    const float* v = (const float*)d_in[1];   // [B, Lv, D] f32
    float* out = (float*)d_out;               // [B, Lv] f32

    char* ws = (char*)d_ws;
    size_t off = 0;
    unsigned short* tn = (unsigned short*)(ws + off); off += (size_t)Bsz * Lt * Dd * 2; // 16 MB
    unsigned short* vn = (unsigned short*)(ws + off); off += (size_t)Bsz * Lv * Dd * 2; // 64 MB
    float* text_acc = (float*)(ws + off); off += (size_t)Bsz * Lt * 4;
    float* colsum   = (float*)(ws + off); off += (size_t)Bsz * Lt * 4;
    float* coeff    = (float*)(ws + off); off += (size_t)Bsz * Lt * 4;
    float* vision   = (float*)(ws + off); off += (size_t)Bsz * Lv * 4;

    // zero atomic accumulators (text_acc + colsum are contiguous)
    hipMemsetAsync(text_acc, 0, (size_t)2 * Bsz * Lt * sizeof(float), stream);

    normalize_kernel<<<(Bsz * Lt) / 8, 256, 0, stream>>>(t, tn, Bsz * Lt);
    normalize_kernel<<<(Bsz * Lv) / 8, 256, 0, stream>>>(v, vn, Bsz * Lv);

    const size_t smem1 = 32 * 1024 * 4 + 128 + 32 * PITCH;           // ~161 KB
    pass1_kernel<<<dim3(Lv / 32, Bsz), 256, smem1, stream>>>(tn, vn, text_acc, colsum);

    select_kernel<<<Bsz, 256, 0, stream>>>(text_acc, colsum, coeff);

    const size_t smem2 = 128 * PITCH + 2 * 16 * PITCH;               // ~163 KB
    pass2_kernel<<<dim3(Lv / 128, Bsz), 256, smem2, stream>>>(tn, vn, coeff, vision);

    minmax_kernel<<<Bsz, 256, 0, stream>>>(vision, out);
}
